// ARRBM_31404800868726
// MI455X (gfx1250) — matrix operand order repaired and checked by reading, not compiled, not run on hardware
//
#include <hip/hip_runtime.h>
#include <hip/hip_bf16.h>

typedef __attribute__((ext_vector_type(16))) _Float16 v16h;
typedef __attribute__((ext_vector_type(8)))  float    v8f;

#define N_VIS     128
#define M_HID     256
#define I_DIM     64
#define WA_STRIDE 136   // f16 per row: 272B rows, 16B-aligned, decorrelates banks
#define AT_STRIDE 65    // f32 per row of atmp: breaks 64-bank conflicts

// LDS layout (bytes)
#define OFF_WA    0                         // 256*136*2 = 69632
#define OFF_AT    69632                     // 256*65*4  = 66560
#define OFF_BIAS  (69632 + 66560)           // 1024
#define OFF_VF    (OFF_BIAS + 1024)         // 512
#define OFF_VH    (OFF_VF + 512)            // 256
#define OFF_RB    (OFF_VH + 256)            // 256
#define LDS_BYTES (OFF_RB + 256)            // 138240

__global__ void arrbm_wmma_kernel(const float* __restrict__ vis,
                                  const float* __restrict__ bias,
                                  const float* __restrict__ weight,
                                  float* __restrict__ out) {
  extern __shared__ char lds[];
  _Float16* wA   = (_Float16*)(lds + OFF_WA);   // [256][WA_STRIDE] f16
  float*    atmp = (float*)   (lds + OFF_AT);   // [256][AT_STRIDE] f32 (no bias)
  float*    sbia = (float*)   (lds + OFF_BIAS); // [256]
  float*    visf = (float*)   (lds + OFF_VF);   // [128]
  _Float16* vish = (_Float16*)(lds + OFF_VH);   // [128]
  float*    rbuf = (float*)   (lds + OFF_RB);   // [64]

  const int b    = blockIdx.x;
  const int tid  = threadIdx.x;     // 0..255
  const int wave = tid >> 5;        // 0..7
  const int lane = tid & 31;

  // ---------------- Stage 0: global -> LDS ----------------
  for (int idx = tid; idx < M_HID * N_VIS; idx += 256) {
    const int m = idx >> 7, k = idx & 127;
    wA[m * WA_STRIDE + k] = (_Float16)weight[idx];
  }
  if (tid < N_VIS) {
    const float v = vis[b * N_VIS + tid];
    visf[tid] = v;
    vish[tid] = (_Float16)v;
  }
  sbia[tid] = bias[tid];
  __syncthreads();

  // ---------------- Stage 1: masked-prefix GEMM via WMMA ----------------
  // atmp[m][i] = sum_k visf[k]*W[m][k]*(k < 2i)   (bias folded in at stage 2)
  {
    const int itile = wave & 3;        // i0 = 16*itile
    const int mgrp  = wave >> 2;       // m-tiles mgrp*8 .. mgrp*8+7
    const int i0    = itile * 16;
    const int lhalf = lane >> 4;       // 0 | 1
    const int l15   = lane & 15;
    const int n_abs = i0 + l15;        // this lane's absolute i column

    // B fragments (Vmask), same K order as A: lanes 0-15 hold K 0-7 & 16-23, lanes 16-31 K 8-15 & 24-31
    v16h bf[4];
    #pragma unroll
    for (int kcix = 0; kcix < 4; ++kcix) {
      const int kb = kcix * 32 + (lhalf ? 8 : 0);
      v16h f;
      #pragma unroll
      for (int j = 0; j < 16; ++j) {
        const int k = kb + (j < 8 ? j : j + 8);
        f[j] = (k < 2 * n_abs) ? vish[k] : (_Float16)0.f;
      }
      bf[kcix] = f;
    }

    for (int t = 0; t < 8; ++t) {
      const int m0   = (mgrp * 8 + t) * 16;
      const int mrow = m0 + l15;

      // Load ALL four A fragments first (8 independent ds_load_b128 in flight),
      // so the WMMA chain below can overlap the tail of the loads.
      v16h af[4];
      #pragma unroll
      for (int kcix = 0; kcix < 4; ++kcix) {
        // A-layout (16-bit 16x32): lanes 0-15: K 0-7 & 16-23; lanes 16-31: K 8-15 & 24-31
        const int kb = kcix * 32 + (lhalf ? 8 : 0);
        const _Float16* rp = &wA[mrow * WA_STRIDE + kb];
        #pragma unroll
        for (int j = 0; j < 8; ++j) af[kcix][j]     = rp[j];       // 16B -> ds_load_b128
        #pragma unroll
        for (int j = 0; j < 8; ++j) af[kcix][8 + j] = rp[16 + j];  // 16B -> ds_load_b128
      }

      v8f acc = {};
      #pragma unroll
      for (int kcix = 0; kcix < 4; ++kcix) {
        acc = __builtin_amdgcn_wmma_f32_16x16x32_f16(
                  false, af[kcix], false, bf[kcix], (short)0, acc, false, false);
      }

      // C layout: lanes 0-15 rows r, lanes 16-31 rows r+8; col = i0 + l15
      const int n = i0 + l15;
      #pragma unroll
      for (int r = 0; r < 8; ++r) {
        const int m = m0 + r + 8 * lhalf;
        atmp[m * AT_STRIDE + n] = acc[r];
      }
    }
  }
  __syncthreads();

  // ---------------- Stage 2: 4 cosines per (m,i), product over m ----------------
  #pragma unroll 1
  for (int j = 0; j < 8; ++j) {
    const int i = wave * 8 + j;
    const float v0 = visf[2 * i], v1 = visf[2 * i + 1];
    float pf = 1.f, p0 = 1.f, pw0 = 1.f, pw1 = 1.f;
    #pragma unroll 2
    for (int c = 0; c < 8; ++c) {
      const int m  = c * 32 + lane;
      const float a  = atmp[m * AT_STRIDE + i] + sbia[m];   // bias folded here
      const float w0 = (float)wA[m * WA_STRIDE + 2 * i];
      const float w1 = (float)wA[m * WA_STRIDE + 2 * i + 1];
      pf  *= __cosf(a + v0 * w0 + v1 * w1);  // full prefix (k < 2i+2)
      p0  *= __cosf(a);                       // configs {0,1,2,5}
      pw0 *= __cosf(a + w0);                  // configs {4,6}
      pw1 *= __cosf(a + w1);                  // configs {3,7}
    }
    #pragma unroll
    for (int s = 16; s > 0; s >>= 1) {
      pf  *= __shfl_xor(pf,  s, 32);
      p0  *= __shfl_xor(p0,  s, 32);
      pw0 *= __shfl_xor(pw0, s, 32);
      pw1 *= __shfl_xor(pw1, s, 32);
    }
    if (lane == 0) {
      const float nrm2 = 4.f * p0 * p0 + 2.f * pw0 * pw0 + 2.f * pw1 * pw1;
      rbuf[i] = pf * rsqrtf(nrm2);            // psi1 / normal
    }
  }
  __syncthreads();

  // ---------------- Stage 3: product over i, Sz filter, write ----------------
  if (wave == 0) {
    float t = rbuf[lane] * rbuf[lane + 32];
    float d = (visf[2 * lane]      - visf[2 * lane + 1]) +
              (visf[2 * lane + 64] - visf[2 * lane + 65]);
    #pragma unroll
    for (int s = 16; s > 0; s >>= 1) {
      t *= __shfl_xor(t, s, 32);
      d += __shfl_xor(d, s, 32);
    }
    if (lane == 0) {
      const float sz = 0.5f * d;              // sum s_alpha - sum s_beta
      out[b] = (sz != 0.f) ? 0.f : t;
    }
  }
}

extern "C" void kernel_launch(void* const* d_in, const int* in_sizes, int n_in,
                              void* d_out, int out_size, void* d_ws, size_t ws_size,
                              hipStream_t stream) {
  const float* vis    = (const float*)d_in[0];  // [1024,128]
  const float* hbias  = (const float*)d_in[1];  // [256]
  const float* weight = (const float*)d_in[2];  // [256,128]
  float* out = (float*)d_out;                   // [1024]
  (void)in_sizes; (void)n_in; (void)d_ws; (void)ws_size;

  hipFuncSetAttribute((const void*)arrbm_wmma_kernel,
                      hipFuncAttributeMaxDynamicSharedMemorySize, LDS_BYTES);
  arrbm_wmma_kernel<<<1024, 256, LDS_BYTES, stream>>>(vis, hbias, weight, out);
}
